// AE_7765300871746
// MI455X (gfx1250) — matrix operand order repaired and checked by reading, not compiled, not run on hardware
//
#include <hip/hip_runtime.h>
#include <hip/hip_bf16.h>

typedef __attribute__((ext_vector_type(16))) _Float16 v16h;
typedef __attribute__((ext_vector_type(8)))  _Float16 v8h;
typedef __attribute__((ext_vector_type(8)))  float    v8f;

#define BATCHN 256
#define SEQN   64
#define FEATN  32
#define INCHN  33
#define NSTEPS 63
#define BT     16
#define ACT_S  72      /* halves per LDS activation row (16B aligned, conflict-friendly) */
#define G_S    1065    /* floats per LDS G-buffer row (covers 1056-wide halves) */

// fast device math: v_rcp_f32 / v_exp_f32 based (no IEEE divide chains on the
// serial critical path; error ~1ulp, far below the f16-operand noise floor)
__device__ __forceinline__ float fast_rcp(float x){ return __builtin_amdgcn_rcpf(x); }
__device__ __forceinline__ float siluf(float x){ return x * fast_rcp(1.0f + __expf(-x)); }
__device__ __forceinline__ float tanh_fast(float x){
  float x2 = fminf(fmaxf(2.0f*x, -30.0f), 30.0f);   // clamp avoids inf*0=NaN
  float e  = __expf(x2);
  return (e - 1.0f) * fast_rcp(e + 1.0f);
}
__device__ __forceinline__ float eluf(float x){ return x > 0.0f ? x : (__expf(x)-1.0f); }

// A fragment (16x32 f16, M=lane&15): elements 0-15 = K[hi*16 .. hi*16+16)
__device__ __forceinline__ v16h loadA(const _Float16* act, int stride, int m, int kbase, int hi){
  const _Float16* p0 = act + m*stride + kbase + hi*16;
  v8h lo = *(const v8h*)p0;
  v8h hh = *(const v8h*)(p0 + 8);
  v16h a;
  #pragma unroll
  for (int i=0;i<8;++i){ a[i]=lo[i]; a[i+8]=hh[i]; }
  return a;
}
// B fragment (32x16 f16, N=lane&15): lanes 0-15 hold K=0..15, lanes 16-31 hold K=16..31.
// Weights pre-transposed to WT[N][K] so this is one contiguous 32-byte load.
__device__ __forceinline__ v16h loadB(const _Float16* WT, int n, int Ktot, int kbase, int hi){
  return *(const v16h*)(WT + (size_t)n*Ktot + kbase + hi*16);
}
#define WMMA_F16(a,b,c) __builtin_amdgcn_wmma_f32_16x16x32_f16(false,(a),false,(b),(short)0,(c),false,false)

// One 16x16 hidden-layer tile per wave: out = silu(in @ W + bias), f16 activations in LDS.
__device__ __forceinline__ void hidden_layer(const _Float16* in, const _Float16* WT, int Ktot,
                                             const float* bias, _Float16* out,
                                             int ntile, int nl, int hi){
  float bs = bias[ntile*16 + nl];
  v8f c = {};
  for (int kc=0; kc<Ktot; kc+=32)
    c = WMMA_F16(loadA(in, ACT_S, nl, kc, hi), loadB(WT, ntile*16+nl, Ktot, kc, hi), c);
  #pragma unroll
  for (int r=0;r<8;++r)
    out[(r + hi*8)*ACT_S + ntile*16 + nl] = (_Float16)siluf(c[r] + bs);
}

// ---------------------------------------------------------------------------
// one-shot helpers
// ---------------------------------------------------------------------------
__global__ void f16t_kernel(const float* __restrict__ src, _Float16* __restrict__ dst, int K, int N){
  int idx = blockIdx.x*blockDim.x + threadIdx.x;
  if (idx >= K*N) return;
  int k = idx / N, n = idx - k*N;
  dst[(size_t)n*K + k] = (_Float16)src[idx];   // WT[n][k] = W[k][n]
}

__global__ void spline_kernel(const float* __restrict__ coeffs,
                              float* __restrict__ D0, float* __restrict__ DH, float* __restrict__ D1){
  int idx = blockIdx.x*blockDim.x + threadIdx.x;
  if (idx >= BATCHN*INCHN) return;
  int b = idx / INCHN, ch = idx - b*INCHN;
  float y[SEQN];
  if (ch == 0){
    for (int i=0;i<SEQN;++i) y[i] = (float)i * (64.0f/63.0f) * (1.0f/64.0f);
  } else {
    for (int i=0;i<SEQN;++i) y[i] = coeffs[((size_t)b*SEQN + i)*FEATN + (ch-1)];
  }
  float cp[62], dp[62], M[64];
  cp[0] = 0.25f;
  dp[0] = 6.0f*(y[2]-2.0f*y[1]+y[0]) * 0.25f;
  for (int k=1;k<62;++k){
    float rk = 6.0f*(y[k+2]-2.0f*y[k+1]+y[k]);
    float m = 1.0f/(4.0f - cp[k-1]);
    cp[k] = m;
    dp[k] = (rk - dp[k-1])*m;
  }
  M[0]=0.f; M[63]=0.f;
  M[62]=dp[61];
  for (int k=60;k>=0;--k) M[k+1] = dp[k] - cp[k]*M[k+2];
  for (int i=0;i<63;++i){
    float bi = (y[i+1]-y[i]) - (2.0f*M[i]+M[i+1])*(1.0f/6.0f);
    float ci = 0.5f*M[i];
    float di = (M[i+1]-M[i])*(1.0f/6.0f);
    size_t o = (size_t)(i*BATCHN + b)*INCHN + ch;
    D0[o] = bi;
    DH[o] = bi + ci + 0.75f*di;
    D1[o] = bi + 2.0f*ci + 3.0f*di;
  }
}

__global__ void init_kernel(const float* __restrict__ coeffs,
                            const float* __restrict__ e0W, const float* __restrict__ e0b,
                            const float* __restrict__ d0W, const float* __restrict__ d0b,
                            float* __restrict__ Z0E, float* __restrict__ Z0D,
                            float* __restrict__ encOut, float* __restrict__ decTraj){
  int idx = blockIdx.x*blockDim.x + threadIdx.x;
  if (idx >= BATCHN*96) return;
  int b = idx/96, o = idx - b*96;
  const float* cb = coeffs + (size_t)b*SEQN*FEATN;  // row s=0 ; X0 = [t(0)=0, coeffs]
  if (o < 64){
    float s = e0b[o];
    #pragma unroll
    for (int k=1;k<33;++k) s += cb[k-1]*e0W[k*64 + o];
    Z0E[b*64+o] = s;
    encOut[((size_t)b*SEQN)*64 + o] = s;
  } else {
    int oo = o - 64;
    float s = d0b[oo];
    #pragma unroll
    for (int k=1;k<33;++k) s += cb[k-1]*d0W[k*32 + oo];
    Z0D[b*32+oo] = s;
    decTraj[((size_t)b*SEQN)*32 + oo] = s;
  }
}

// ---------------------------------------------------------------------------
// persistent RK4 scan: one workgroup = one 16-row batch tile, full 63 steps
// ---------------------------------------------------------------------------
// LDS offsets (in halves) for the six small hidden-layer weight matrices
#define sWE0   0        /* [64][64]  */
#define sWEH0  4096     /* [64][64]  */
#define sWEH1  8192     /* [64][64]  */
#define sWD0   12288    /* [64][32]  */
#define sWDH0  14336    /* [64][64]  */
#define sWDH1  18432    /* [64][64]  */
#define WH_TOT 22528

__global__ __launch_bounds__(256) void scan_kernel(
    const _Float16* __restrict__ WE0T, const _Float16* __restrict__ WEH0T,
    const _Float16* __restrict__ WEH1T, const _Float16* __restrict__ WEFT,
    const _Float16* __restrict__ WD0T, const _Float16* __restrict__ WDH0T,
    const _Float16* __restrict__ WDH1T, const _Float16* __restrict__ WDFT,
    const float* __restrict__ feb0, const float* __restrict__ febh, const float* __restrict__ febf,
    const float* __restrict__ fdb0, const float* __restrict__ fdbh, const float* __restrict__ fdbf,
    const float* __restrict__ D0g, const float* __restrict__ DHg, const float* __restrict__ D1g,
    const float* __restrict__ Z0E, const float* __restrict__ Z0D,
    float* __restrict__ encOut, float* __restrict__ decTraj)
{
  __shared__ __align__(16) float    G[16*G_S];
  __shared__ __align__(16) _Float16 WH[WH_TOT];      // LDS-resident hidden-layer weights
  __shared__ __align__(16) _Float16 actE[16*ACT_S];
  __shared__ __align__(16) _Float16 actA[16*ACT_S];
  __shared__ __align__(16) _Float16 actB[16*ACT_S];
  __shared__ __align__(16) _Float16 actD[16*ACT_S];
  __shared__ float baseE[16*64], accE[16*64], dzE[16*64];
  __shared__ float baseD[16*32], accD[16*32], dzD[16*32];
  __shared__ float dx0[16*INCHN], dxh[16*INCHN], dx1[16*INCHN];

  const int tid  = threadIdx.x;
  const int lane = tid & 31;
  const int wave = tid >> 5;
  const int nl   = lane & 15;
  const int hi   = lane >> 4;
  const int b0   = blockIdx.x * BT;

  // stage small weights into LDS once (serial-path latency killer otherwise)
  for (int i=tid;i<4096;i+=256) WH[sWE0 +i] = WE0T [i];
  for (int i=tid;i<4096;i+=256) WH[sWEH0+i] = WEH0T[i];
  for (int i=tid;i<4096;i+=256) WH[sWEH1+i] = WEH1T[i];
  for (int i=tid;i<2048;i+=256) WH[sWD0 +i] = WD0T [i];
  for (int i=tid;i<4096;i+=256) WH[sWDH0+i] = WDH0T[i];
  for (int i=tid;i<4096;i+=256) WH[sWDH1+i] = WDH1T[i];

  for (int i=tid;i<16*64;i+=256){ baseE[i]=Z0E[b0*64+i]; accE[i]=0.f; dzE[i]=0.f; }
  for (int i=tid;i<16*32;i+=256){ baseD[i]=Z0D[b0*32+i]; accD[i]=0.f; dzD[i]=0.f; }
  __syncthreads();

  for (int step=0; step<NSTEPS; ++step){
    {
      const size_t off = (size_t)(step*BATCHN + b0)*INCHN;
      const float* p0 = D0g + off; const float* ph = DHg + off; const float* p1 = D1g + off;
      for (int i=tid;i<16*INCHN;i+=256){ dx0[i]=p0[i]; dxh[i]=ph[i]; dx1[i]=p1[i]; }
    }
    __syncthreads();

    for (int st=0; st<4; ++st){
      const float  cIn  = (st==0) ? 0.f : ((st==3) ? 1.f : 0.5f);
      const float  wAcc = (st==1 || st==2) ? 2.f : 1.f;
      const float* dxs  = (st==0) ? dx0 : ((st==3) ? dx1 : dxh);

      // stage-input states -> f16 A operands
      for (int i=tid;i<16*64;i+=256) actE[(i>>6)*ACT_S + (i&63)] = (_Float16)(baseE[i] + cIn*dzE[i]);
      for (int i=tid;i<16*32;i+=256) actD[(i>>5)*ACT_S + (i&31)] = (_Float16)(baseD[i] + cIn*dzD[i]);
      __syncthreads();

      // ---- encoder field MLP ----
      if (wave<4) hidden_layer(actE, WH+sWE0 , 64, feb0    , actA, wave, nl, hi);
      __syncthreads();
      if (wave<4) hidden_layer(actA, WH+sWEH0, 64, febh    , actB, wave, nl, hi);
      __syncthreads();
      if (wave<4) hidden_layer(actB, WH+sWEH1, 64, febh+64 , actA, wave, nl, hi);
      __syncthreads();
      // final layer (2112 cols) in two 1056-wide halves, tanh into G, then reduce vs dX
      for (int half=0; half<2; ++half){
        const int colBase = half*1056;
        for (int t=wave; t<66; t+=8){
          int cg = colBase + t*16 + nl;
          float bs = febf[cg];
          v8f c = {};
          #pragma unroll
          for (int kc=0;kc<64;kc+=32)
            c = WMMA_F16(loadA(actA, ACT_S, nl, kc, hi), loadB(WEFT, cg, 64, kc, hi), c);
          #pragma unroll
          for (int r=0;r<8;++r) G[(r + hi*8)*G_S + t*16 + nl] = tanh_fast(c[r] + bs);
        }
        __syncthreads();
        const int hBase = half*32;
        for (int p=tid;p<512;p+=256){
          int bt = p>>5, hL = p&31;
          const float* g  = &G[bt*G_S + hL*33];
          const float* dd = &dxs[bt*INCHN];
          float s = 0.f;
          #pragma unroll
          for (int i=0;i<33;++i) s += g[i]*dd[i];
          int h = hBase + hL;
          dzE[bt*64+h] = s;
          accE[bt*64+h] += wAcc*s;
        }
        __syncthreads();
      }

      // ---- decoder field MLP ----
      if (wave<4) hidden_layer(actD, WH+sWD0 , 32, fdb0    , actB, wave, nl, hi);
      __syncthreads();
      if (wave<4) hidden_layer(actB, WH+sWDH0, 64, fdbh    , actE, wave, nl, hi);
      __syncthreads();
      if (wave<4) hidden_layer(actE, WH+sWDH1, 64, fdbh+64 , actB, wave, nl, hi);
      __syncthreads();
      for (int half=0; half<2; ++half){
        const int colBase = half*1040;
        for (int t=wave; t<65; t+=8){
          int cg = colBase + t*16 + nl;
          float bs = fdbf[cg];
          v8f c = {};
          #pragma unroll
          for (int kc=0;kc<64;kc+=32)
            c = WMMA_F16(loadA(actB, ACT_S, nl, kc, hi), loadB(WDFT, cg, 64, kc, hi), c);
          #pragma unroll
          for (int r=0;r<8;++r) G[(r + hi*8)*G_S + t*16 + nl] = tanh_fast(c[r] + bs);
        }
        __syncthreads();
        const int oBase = half*16;
        {
          int p = tid;  // exactly 256 (b,o) pairs per half
          int bt = p>>4, oL = p&15;
          const float* g = &G[bt*G_S + oL*65];
          float s = g[0]*dxs[bt*INCHN];            // dctrl[0] = dX_t
          #pragma unroll
          for (int j=1;j<65;++j) s += g[j]*dzE[bt*64 + (j-1)];
          int o = oBase + oL;
          dzD[bt*32+o] = s;
          accD[bt*32+o] += wAcc*s;
        }
        __syncthreads();
      }
    } // RK4 stages

    // state update + trajectory write-out
    for (int i=tid;i<16*64;i+=256){
      float v = baseE[i] + accE[i]*(1.0f/6.0f);
      baseE[i] = v; accE[i] = 0.f;
      int bt = i>>6, h = i&63;
      encOut[((size_t)(b0+bt)*SEQN + (step+1))*64 + h] = v;
    }
    for (int i=tid;i<16*32;i+=256){
      float v = baseD[i] + accD[i]*(1.0f/6.0f);
      baseD[i] = v; accD[i] = 0.f;
      int bt = i>>5, o = i&31;
      decTraj[((size_t)(b0+bt)*SEQN + (step+1))*32 + o] = v;
    }
    __syncthreads();
  }
}

// ---------------------------------------------------------------------------
// readout: pred = elu(decode @ ro_W0 + b0) @ ro_W1 + b1  — one wave per M-tile
// ---------------------------------------------------------------------------
__global__ __launch_bounds__(32) void readout_kernel(
    const float* __restrict__ decTraj,
    const _Float16* __restrict__ RO0T, const _Float16* __restrict__ RO1T,
    const float* __restrict__ rob0, const float* __restrict__ rob1,
    float* __restrict__ pred)
{
  __shared__ __align__(16) _Float16 dIn[16*40];
  __shared__ __align__(16) _Float16 hbuf[16*136];
  const int lane = threadIdx.x;
  const int nl = lane & 15, hi = lane >> 4;
  const size_t r0 = (size_t)blockIdx.x * 16;

  for (int i=lane;i<16*32;i+=32){
    int row = i>>5, c = i&31;
    dIn[row*40 + c] = (_Float16)decTraj[(r0+row)*32 + c];
  }
  __syncthreads();

  #pragma unroll
  for (int t=0;t<8;++t){
    float bs = rob0[t*16+nl];
    v8f c = {};
    c = WMMA_F16(loadA(dIn, 40, nl, 0, hi), loadB(RO0T, t*16+nl, 32, 0, hi), c);
    #pragma unroll
    for (int r=0;r<8;++r) hbuf[(r + hi*8)*136 + t*16 + nl] = (_Float16)eluf(c[r] + bs);
  }
  __syncthreads();

  #pragma unroll
  for (int t=0;t<2;++t){
    float bs = rob1[t*16+nl];
    v8f c = {};
    #pragma unroll
    for (int kc=0;kc<128;kc+=32)
      c = WMMA_F16(loadA(hbuf, 136, nl, kc, hi), loadB(RO1T, t*16+nl, 128, kc, hi), c);
    #pragma unroll
    for (int r=0;r<8;++r) pred[(r0 + r + hi*8)*32 + t*16 + nl] = c[r] + bs;
  }
}

// ---------------------------------------------------------------------------
// workspace layout (bytes, each section 256-aligned)
// ---------------------------------------------------------------------------
enum : size_t {
  OFF_WE0T  = 0,                       // [64][64] f16
  OFF_WEH0T = 8192,
  OFF_WEH1T = 16384,
  OFF_WEFT  = 24576,                   // [2112][64] f16
  OFF_WD0T  = 294912,                  // [64][32] f16
  OFF_WDH0T = 299008,
  OFF_WDH1T = 307200,
  OFF_WDFT  = 315392,                  // [2080][64] f16
  OFF_RO0T  = 581632,                  // [128][32] f16
  OFF_RO1T  = 589824,                  // [32][128] f16
  OFF_D0    = 598016,                  // [63][256][33] f32
  OFF_DH    = OFF_D0 + 2128896,
  OFF_D1    = OFF_DH + 2128896,
  OFF_Z0E   = OFF_D1 + 2128896,        // [256][64] f32
  OFF_Z0D   = OFF_Z0E + 65536,         // [256][32] f32
  OFF_DEC   = OFF_Z0D + 32768,         // [256][64][32] f32
};

extern "C" void kernel_launch(void* const* d_in, const int* in_sizes, int n_in,
                              void* d_out, int out_size, void* d_ws, size_t ws_size,
                              hipStream_t stream) {
  (void)in_sizes; (void)n_in; (void)out_size; (void)ws_size;
  const float* coeffs = (const float*)d_in[0];
  const float* e0W = (const float*)d_in[1];
  const float* e0b = (const float*)d_in[2];
  const float* d0W = (const float*)d_in[3];
  const float* d0b = (const float*)d_in[4];
  const float* feW0 = (const float*)d_in[5];
  const float* feb0 = (const float*)d_in[6];
  const float* feWh = (const float*)d_in[7];
  const float* febh = (const float*)d_in[8];
  const float* feWf = (const float*)d_in[9];
  const float* febf = (const float*)d_in[10];
  const float* fdW0 = (const float*)d_in[11];
  const float* fdb0 = (const float*)d_in[12];
  const float* fdWh = (const float*)d_in[13];
  const float* fdbh = (const float*)d_in[14];
  const float* fdWf = (const float*)d_in[15];
  const float* fdbf = (const float*)d_in[16];
  const float* roW0 = (const float*)d_in[17];
  const float* rob0 = (const float*)d_in[18];
  const float* roW1 = (const float*)d_in[19];
  const float* rob1 = (const float*)d_in[20];

  char* ws = (char*)d_ws;
  _Float16* WE0T  = (_Float16*)(ws + OFF_WE0T);
  _Float16* WEH0T = (_Float16*)(ws + OFF_WEH0T);
  _Float16* WEH1T = (_Float16*)(ws + OFF_WEH1T);
  _Float16* WEFT  = (_Float16*)(ws + OFF_WEFT);
  _Float16* WD0T  = (_Float16*)(ws + OFF_WD0T);
  _Float16* WDH0T = (_Float16*)(ws + OFF_WDH0T);
  _Float16* WDH1T = (_Float16*)(ws + OFF_WDH1T);
  _Float16* WDFT  = (_Float16*)(ws + OFF_WDFT);
  _Float16* RO0T  = (_Float16*)(ws + OFF_RO0T);
  _Float16* RO1T  = (_Float16*)(ws + OFF_RO1T);
  float* D0  = (float*)(ws + OFF_D0);
  float* DH  = (float*)(ws + OFF_DH);
  float* D1  = (float*)(ws + OFF_D1);
  float* Z0E = (float*)(ws + OFF_Z0E);
  float* Z0D = (float*)(ws + OFF_Z0D);
  float* DEC = (float*)(ws + OFF_DEC);

  float* pred = (float*)d_out;                  // [256,64,32]
  float* enc  = (float*)d_out + 256*64*32;      // [256,64,64]

  auto T = [&](const float* src, _Float16* dst, int K, int N){
    int tot = K*N;
    f16t_kernel<<<(tot+255)/256, 256, 0, stream>>>(src, dst, K, N);
  };
  T(feW0,        WE0T , 64, 64);
  T(feWh,        WEH0T, 64, 64);
  T(feWh + 4096, WEH1T, 64, 64);
  T(feWf,        WEFT , 64, 2112);
  T(fdW0,        WD0T , 32, 64);
  T(fdWh,        WDH0T, 64, 64);
  T(fdWh + 4096, WDH1T, 64, 64);
  T(fdWf,        WDFT , 64, 2080);
  T(roW0,        RO0T , 32, 128);
  T(roW1,        RO1T , 128, 32);

  spline_kernel<<<(BATCHN*INCHN + 255)/256, 256, 0, stream>>>(coeffs, D0, DH, D1);
  init_kernel<<<(BATCHN*96 + 255)/256, 256, 0, stream>>>(coeffs, e0W, e0b, d0W, d0b,
                                                         Z0E, Z0D, enc, DEC);
  scan_kernel<<<BATCHN/BT, 256, 0, stream>>>(WE0T, WEH0T, WEH1T, WEFT,
                                             WD0T, WDH0T, WDH1T, WDFT,
                                             feb0, febh, febf, fdb0, fdbh, fdbf,
                                             D0, DH, D1, Z0E, Z0D, enc, DEC);
  readout_kernel<<<BATCHN*SEQN/16, 32, 0, stream>>>(DEC, RO0T, RO1T, rob0, rob1, pred);
}
